// STblock_30966714204615
// MI455X (gfx1250) — matrix operand order repaired and checked by reading, not compiled, not run on hardware
//
#include <hip/hip_runtime.h>

typedef __attribute__((ext_vector_type(16))) _Float16 v16h;
typedef __attribute__((ext_vector_type(8)))  _Float16 v8h;
typedef __attribute__((ext_vector_type(8)))  float    v8f;

#define D_ 128

__device__ __forceinline__ float gelu_f(float x) {
  return 0.5f * x * (1.0f + erff(x * 0.70710678118654752f));
}

// ---------------------------------------------------------------------------
// Generic batched WMMA GEMM: Y[z] = act( beta*Y[z] + A[z] @ W[z] + bias )
// A: [M,K] row-major, W: [K,N] row-major, Y: [M,N].  128 threads = 4 waves,
// each wave owns a 16x64 strip of a 64x64 tile; K stepped by 32 through LDS
// (f32 -> f16 on the way in).  LDS layouts are fragment-native:
//   Al[row][k]  : A-lane fragment = two contiguous 16B runs  -> ds_load_b128
//   Blt[n][k]   : B tile stored transposed; lane fragment = two separate
//                 16B runs (two ds_load_b128)
// Rows padded to 40 halves (80B) to avoid 64B bank-stride conflicts.
// ---------------------------------------------------------------------------
__global__ void gemm_k(const float* __restrict__ A, const float* __restrict__ W,
                       const float* __restrict__ bias, float* __restrict__ Y,
                       int M, int N, int K, long sA, long sW, long sY,
                       int act, int beta)
{
  __shared__ __align__(16) _Float16 Al[64][40];
  __shared__ __align__(16) _Float16 Blt[64][40];
  const float* Ab = A + (long)blockIdx.z * sA;
  const float* Wb = W + (long)blockIdx.z * sW;
  float*       Yb = Y + (long)blockIdx.z * sY;
  const int bm = blockIdx.y * 64, bn = blockIdx.x * 64;
  const int tid = threadIdx.x, wave = tid >> 5, lane = tid & 31;
  const int half = lane >> 4, l16 = lane & 15;
  const bool k4 = (K & 3) == 0;
  const bool n4 = (N & 3) == 0;
  v8f acc[4] = {};
  const int kt_end = (K + 31) >> 5;
  for (int kt = 0; kt < kt_end; ++kt) {
    const int k0 = kt << 5;
    // ---- stage A tile: 64x32, float4 fast path -> global_load_b128 ----
    for (int g = tid; g < (64 * 32) / 4; g += 128) {
      int base = g * 4;
      int r = base >> 5, c = base & 31;
      int gm = bm + r, gk = k0 + c;
      if (k4 && gm < M && gk + 3 < K) {
        const float4 v4 = *(const float4*)(Ab + (long)gm * K + gk);
        Al[r][c]     = (_Float16)v4.x;
        Al[r][c + 1] = (_Float16)v4.y;
        Al[r][c + 2] = (_Float16)v4.z;
        Al[r][c + 3] = (_Float16)v4.w;
      } else {
#pragma unroll
        for (int j = 0; j < 4; ++j)
          Al[r][c + j] = (gm < M && gk + j < K)
                             ? (_Float16)Ab[(long)gm * K + gk + j] : (_Float16)0.0f;
      }
    }
    // ---- stage B tile transposed: Blt[n][k] ----
    for (int g = tid; g < (32 * 64) / 4; g += 128) {
      int base = g * 4;
      int r = base >> 6, c = base & 63;            // r = k-local, c = n-local
      int gk = k0 + r, gn = bn + c;
      if (n4 && gk < K && gn + 3 < N) {
        const float4 v4 = *(const float4*)(Wb + (long)gk * N + gn);
        Blt[c][r]     = (_Float16)v4.x;
        Blt[c + 1][r] = (_Float16)v4.y;
        Blt[c + 2][r] = (_Float16)v4.z;
        Blt[c + 3][r] = (_Float16)v4.w;
      } else {
#pragma unroll
        for (int j = 0; j < 4; ++j)
          Blt[c + j][r] = (gk < K && gn + j < N)
                              ? (_Float16)Wb[(long)gk * N + gn + j] : (_Float16)0.0f;
      }
    }
    // prefetch next K tile (global_prefetch_b8; speculative, OOB-safe)
    if (kt + 1 < kt_end) {
      int pr = bm + (tid & 63); if (pr >= M) pr = M - 1;
      __builtin_prefetch(Ab + (long)pr * K + (k0 + 32), 0, 1);
      int pk = k0 + 32 + (tid & 31); if (pk >= K) pk = K - 1;
      __builtin_prefetch(Wb + (long)pk * N + bn, 0, 1);
    }
    __syncthreads();
    // ---- A fragment: two contiguous 16B runs of this lane's row ----
    const int am = wave * 16 + l16;
    const v8h* arow = (const v8h*)(&Al[am][0]);
    v8h a_lo = arow[half];        // K = 8*half .. 8*half+7
    v8h a_hi = arow[2 + half];    // K = 16+8*half .. +7
    v16h af = __builtin_shufflevector(a_lo, a_hi,
        0, 1, 2, 3, 4, 5, 6, 7, 8, 9, 10, 11, 12, 13, 14, 15);
#pragma unroll
    for (int nt = 0; nt < 4; ++nt) {
      const v8h* brow = (const v8h*)(&Blt[nt * 16 + l16][0]);
      v8h b_lo = brow[half];            // K = 8*half .. 8*half+7
      v8h b_hi = brow[2 + half];        // K = 16+8*half .. +7
      v16h bf = __builtin_shufflevector(b_lo, b_hi,
          0, 1, 2, 3, 4, 5, 6, 7, 8, 9, 10, 11, 12, 13, 14, 15);
      acc[nt] = __builtin_amdgcn_wmma_f32_16x16x32_f16(
          false, af, false, bf, (short)0, acc[nt], false, false);
    }
    __syncthreads();
  }
#pragma unroll
  for (int nt = 0; nt < 4; ++nt) {
#pragma unroll
    for (int r = 0; r < 8; ++r) {
      int gm = bm + wave * 16 + half * 8 + r;   // C/D 16x16 f32 layout
      int gn = bn + nt * 16 + l16;
      if (gm < M && gn < N) {
        float v = acc[nt][r];
        if (beta) v += Yb[(long)gm * N + gn];
        if (bias) v += bias[gn];
        if (act == 1)      v = gelu_f(v);
        else if (act == 2) v = 1.0f / (1.0f + expf(-v));
        Yb[(long)gm * N + gn] = v;
      }
    }
  }
}

static inline void gemm(hipStream_t st, const float* A, const float* W, const float* bias,
                        float* Y, int M, int N, int K, int batch,
                        long sA, long sW, long sY, int act, int beta) {
  dim3 g((N + 63) / 64, (M + 63) / 64, batch);
  gemm_k<<<g, 128, 0, st>>>(A, W, bias, Y, M, N, K, sA, sW, sY, act, beta);
}

// ---------------------------------------------------------------------------
// DFT matrices: cosF/sinF [F,T] (rfft), icosT/isinT [T,F] (irfft with scaling)
// ---------------------------------------------------------------------------
__global__ void gen_dft_k(float* cosF, float* sinF, float* icosT, float* isinT,
                          int T, int F) {
  int i = blockIdx.x * blockDim.x + threadIdx.x;
  if (i >= F * T) return;
  int f = i / T, t = i % T;
  float ang = 6.283185307179586f * (float)(f * t) / (float)T;
  float c = cosf(ang), s = sinf(ang);
  cosF[i] = c;
  sinF[i] = -s;                                    // Im part of rfft
  float scale = (f == 0 || 2 * f == T) ? (1.0f / T) : (2.0f / T);
  icosT[t * F + f] = scale * c;
  isinT[t * F + f] = -scale * s;                   // x += -scale*sin * Im
}

__global__ void amp_k(const float* re, const float* im, float* amp,
                      int n, int F, int Dm) {
  int i = blockIdx.x * blockDim.x + threadIdx.x;
  if (i >= n) return;
  int f = (i / Dm) % F;
  amp[i] = (f == 0) ? 0.0f : sqrtf(re[i] * re[i] + im[i] * im[i]);
}

// per-(b,d): 5th largest amplitude over freqs
__global__ void thr_k(const float* amp, float* thr, int F, int Dm, int Bc) {
  int i = blockIdx.x * blockDim.x + threadIdx.x;
  if (i >= Bc * Dm) return;
  int b = i / Dm, d = i % Dm;
  float t5[5] = {-1.f, -1.f, -1.f, -1.f, -1.f};
  for (int f = 0; f < F; ++f) {
    float a = amp[((long)b * F + f) * Dm + d];
    if (a > t5[4]) {
      t5[4] = a;
#pragma unroll
      for (int j = 4; j > 0; --j)
        if (t5[j] > t5[j - 1]) { float tt = t5[j]; t5[j] = t5[j - 1]; t5[j - 1] = tt; }
    }
  }
  thr[i] = t5[4];
}

__global__ void mask_k(float* amp, const float* thr, float* re, float* im,
                       int n, int F, int Dm) {
  int i = blockIdx.x * blockDim.x + threadIdx.x;
  if (i >= n) return;
  int d = i % Dm, b = i / (F * Dm);
  if (amp[i] < thr[b * Dm + d]) { re[i] = 0.f; im[i] = 0.f; amp[i] = 0.f; }
}

// one block: mean amplitude per freq -> top-2 freq indices -> periods (device)
// dims: [idx0, idx1, per0, per1, nb0, nb1, Tp0, Tp1]
__global__ void period_k(const float* amp, int* dims, int Bc, int F, int Dm, int T) {
  __shared__ float am[256];
  int f = threadIdx.x;
  float s = 0.f;
  if (f < F)
    for (int b = 0; b < Bc; ++b)
      for (int d = 0; d < Dm; ++d)
        s += amp[((long)b * F + f) * Dm + d];
  am[f] = s;
  __syncthreads();
  if (threadIdx.x == 0) {
    int j0 = 1, j1 = 2; float b0 = -1.f, b1 = -1.f;
    for (int ff = 1; ff < F; ++ff) {
      float v = am[ff];
      if (v > b0)      { b1 = b0; j1 = j0; b0 = v; j0 = ff; }
      else if (v > b1) { b1 = v; j1 = ff; }
    }
    dims[0] = j0; dims[1] = j1;
    for (int i = 0; i < 2; ++i) {
      int idx = dims[i]; if (idx < 1) idx = 1;
      int per = T / idx; if (per < 1) per = 1;
      int nb = (T + per - 1) / per;
      dims[2 + i] = per; dims[4 + i] = nb; dims[6 + i] = nb * per;
    }
  }
}

// per-batch softmax over amplitude at the two selected freqs
__global__ void pw_k(const float* amp, const int* dims, float* wts, int F, int Bc) {
  int b = threadIdx.x;
  if (b >= Bc) return;
  int f0 = dims[0], f1 = dims[1];
  float p0 = 0.f, p1 = 0.f;
  for (int d = 0; d < D_; ++d) {
    p0 += amp[((long)b * F + f0) * D_ + d];
    p1 += amp[((long)b * F + f1) * D_ + d];
  }
  p0 /= D_; p1 /= D_;
  float m = fmaxf(p0, p1);
  float e0 = expf(p0 - m), e1 = expf(p1 - m), es = e0 + e1;
  wts[b * 2] = e0 / es; wts[b * 2 + 1] = e1 / es;
}

// -------------------- elementwise helpers --------------------
__global__ void fill0_k(float* a, int n) {
  int i = blockIdx.x * blockDim.x + threadIdx.x; if (i < n) a[i] = 0.f;
}
__global__ void sub_k(const float* a, const float* b, float* c, int n) {
  int i = blockIdx.x * blockDim.x + threadIdx.x; if (i < n) c[i] = a[i] - b[i];
}
__global__ void add_k(const float* a, const float* b, float* c, int n) {
  int i = blockIdx.x * blockDim.x + threadIdx.x; if (i < n) c[i] = a[i] + b[i];
}
__global__ void sout_k(const float* res, const float* x, float* so, int n) {
  int i = blockIdx.x * blockDim.x + threadIdx.x; if (i < n) so[i] = res[i] + 2.f * x[i];
}

// pad season to Tp (runtime, from dims) rows per batch
__global__ void pad_k(const float* x, float* xp, const int* dims, int which,
                      int T, int Bc) {
  int Tp = dims[6 + which];
  int i = blockIdx.x * blockDim.x + threadIdx.x;
  if (i >= Bc * 2 * T * D_) return;
  int d = i % D_, t = (i / D_) % (2 * T), b = i / (D_ * 2 * T);
  if (t >= Tp) return;
  xp[((long)b * Tp + t) * D_ + d] = (t < T) ? x[((long)b * T + t) * D_ + d] : 0.f;
}

// fused MCDC: (conv1x1 + b1 + conv3x3(pad=1) + b3)/2, spatial = (nb,per) runtime
__global__ void mcdc_k(const float* __restrict__ X, const float* __restrict__ w1,
                       const float* __restrict__ b1, const float* __restrict__ w3,
                       const float* __restrict__ b3, float* __restrict__ Y,
                       const int* __restrict__ dims, int which,
                       int Bc, int Cin, int Cout, int dogelu)
{
  int per = dims[2 + which], nb = dims[4 + which];
  int Tp = per * nb;
  int i = blockIdx.x * blockDim.x + threadIdx.x;
  long total = (long)Bc * Tp * Cout;
  if (i >= total) return;
  int co = i % Cout, t = (i / Cout) % Tp, b = i / (Cout * Tp);
  int h = t / per, w = t % per;
  const float* xb = X + (long)b * Tp * Cin;
  float acc = 0.f;
  for (int ci = 0; ci < Cin; ++ci) {
    acc += w1[co * Cin + ci] * xb[(long)t * Cin + ci];
    const float* wk = w3 + ((long)co * Cin + ci) * 9;
#pragma unroll
    for (int dh = -1; dh <= 1; ++dh) {
      int hh = h + dh; if (hh < 0 || hh >= nb) continue;
#pragma unroll
      for (int dw = -1; dw <= 1; ++dw) {
        int ww = w + dw; if (ww < 0 || ww >= per) continue;
        acc += wk[(dh + 1) * 3 + (dw + 1)] * xb[(long)(hh * per + ww) * Cin + ci];
      }
    }
  }
  float v = 0.5f * (acc + b1[co] + b3[co]);
  if (dogelu) v = gelu_f(v);
  Y[(long)b * Tp * Cout + (long)t * Cout + co] = v;
}

// res[b,t,:] += wts[b,which] * y[b, t(<T of Tp), :]
__global__ void accum_k(const float* y, const float* wts, float* res,
                        const int* dims, int which, int T, int Bc) {
  int i = blockIdx.x * blockDim.x + threadIdx.x;
  if (i >= Bc * T * D_) return;
  int d = i % D_, t = (i / D_) % T, b = i / (D_ * T);
  int Tp = dims[6 + which];
  res[i] += wts[b * 2 + which] * y[((long)b * Tp + t) * D_ + d];
}

// scale feats: [mean | per ps: (globalmin, mean-of-patch-std, globalmax)]
__global__ void feats_k(const float* x, float* feats, int T, int Bc) {
  int i = blockIdx.x * blockDim.x + threadIdx.x;
  if (i >= Bc * D_) return;
  int b = i / D_, d = i % D_;
  const float* xb = x + (long)b * T * D_ + d;
  float mn = 3.4e38f, mx = -3.4e38f, sm = 0.f;
  for (int t = 0; t < T; ++t) {
    float v = xb[(long)t * D_];
    sm += v; mn = fminf(mn, v); mx = fmaxf(mx, v);
  }
  feats[b * 1280 + d] = sm / T;
  int pss[3] = {8, 16, 32};
  for (int psi = 0; psi < 3; ++psi) {
    int ps = pss[psi], np = T / ps;
    float ssum = 0.f;
    for (int pn = 0; pn < np; ++pn) {
      float s1 = 0.f, s2 = 0.f;
      for (int j = 0; j < ps; ++j) {
        float v = xb[(long)(pn * ps + j) * D_];
        s1 += v; s2 += v * v;
      }
      float mu = s1 / ps;
      float var = (s2 - ps * mu * mu) / (ps - 1);
      ssum += sqrtf(fmaxf(var, 0.f));
    }
    int base = b * 1280 + 128 + psi * 384;
    feats[base + d]       = mn;
    feats[base + 128 + d] = ssum / np;
    feats[base + 256 + d] = mx;
  }
}

__global__ void ctx_k(const float* feats, float* ctx, int Bc) {
  int i = blockIdx.x * blockDim.x + threadIdx.x;
  if (i < Bc * D_) ctx[i] = feats[(i / D_) * 1280 + (i % D_)];
}

// top-2-of-3 softmax gating
__global__ void moew_k(const float* logits, float* wexp, int Bc) {
  int b = threadIdx.x;
  if (b >= Bc) return;
  float l[3] = {logits[b * 3], logits[b * 3 + 1], logits[b * 3 + 2]};
  int i0 = 0;
  if (l[1] > l[i0]) i0 = 1;
  if (l[2] > l[i0]) i0 = 2;
  int i1 = -1; float v1 = -3.4e38f;
  for (int e = 0; e < 3; ++e)
    if (e != i0 && l[e] > v1) { v1 = l[e]; i1 = e; }
  float e1 = expf(v1 - l[i0]);
  float g0 = 1.f / (1.f + e1), g1 = e1 / (1.f + e1);
  for (int e = 0; e < 3; ++e)
    wexp[b * 3 + e] = (e == i0) ? g0 : (e == i1) ? g1 : 0.f;
}

__global__ void pmean_k(const float* x, float* pm, int T, int ps, int Bc) {
  int Np = T / ps;
  int i = blockIdx.x * blockDim.x + threadIdx.x;
  if (i >= Bc * Np * D_) return;
  int d = i % D_, pn = (i / D_) % Np, b = i / (D_ * Np);
  float s = 0.f;
  for (int j = 0; j < ps; ++j) s += x[((long)b * T + pn * ps + j) * D_ + d];
  pm[i] = s / ps;
}

// moe += wexp[b,e] * ( wg[b]*intra + (1-wg[b])*inter[b,t/ps,:] + x )
__global__ void fuse_k(const float* intra, const float* inter, const float* x,
                       const float* wg, const float* wexp, float* moe,
                       int T, int ps, int e, int Bc) {
  int i = blockIdx.x * blockDim.x + threadIdx.x;
  if (i >= Bc * T * D_) return;
  int d = i % D_, t = (i / D_) % T, b = i / (D_ * T);
  int Np = T / ps, pn = t / ps;
  float wgb = wg[b], we = wexp[b * 3 + e];
  float val = wgb * intra[i] + (1.f - wgb) * inter[((long)b * Np + pn) * D_ + d] + x[i];
  moe[i] += we * val;
}

// layernorm(moe + 2*trend)
__global__ void ln_k(const float* moe, const float* trend, const float* g,
                     const float* bb, float* out, int T, int Bc) {
  int i = blockIdx.x * blockDim.x + threadIdx.x;
  if (i >= Bc * T) return;
  const float* m  = moe   + (long)i * D_;
  const float* tr = trend + (long)i * D_;
  float s = 0.f;
  for (int d = 0; d < D_; ++d) s += m[d] + 2.f * tr[d];
  float mu = s / D_;
  float vv = 0.f;
  for (int d = 0; d < D_; ++d) { float u = m[d] + 2.f * tr[d] - mu; vv += u * u; }
  float r = rsqrtf(vv / D_ + 1e-5f);
  for (int d = 0; d < D_; ++d)
    out[(long)i * D_ + d] = (m[d] + 2.f * tr[d] - mu) * r * g[d] + bb[d];
}

// [B,R,C] -> [B,C,R]
__global__ void tr_k(const float* X, float* Y, int R, int C, int Bc) {
  int i = blockIdx.x * blockDim.x + threadIdx.x;
  if (i >= Bc * R * C) return;
  int c = i % C, r = (i / C) % R, b = i / (R * C);
  Y[((long)b * C + c) * R + r] = X[i];
}

// ---------------------------------------------------------------------------
extern "C" void kernel_launch(void* const* d_in, const int* in_sizes, int n_in,
                              void* d_out, int out_size, void* d_ws, size_t ws_size,
                              hipStream_t stream) {
  (void)in_sizes; (void)out_size; (void)ws_size;
  if (n_in < 65) return;
  auto in = [&](int i) { return (const float*)d_in[i]; };
  const float* X[3] = { in(0), in(1), in(2) };
  const int Ts[3] = {256, 128, 64};
  int q = 3;
  // itcs layer0: conv1[1x1]{w,b}, conv1[3x3]{w,b}, conv2[1x1]{w,b}, conv2[3x3]{w,b}
  const float *c1w0 = in(q++), *c1b0 = in(q++), *c1w3 = in(q++), *c1b3 = in(q++);
  const float *c2w0 = in(q++), *c2b0 = in(q++), *c2w3 = in(q++), *c2b3 = in(q++);
  // ipms layer0
  const float *g1w = in(q++), *g1b = in(q++), *g2w = in(q++), *g2b = in(q++);
  const float* ew[3][12]; // per expert: intra1 w,b  intra2 w,b  inter1 w,b  inter2 w,b  wg1 w,b  wg2 w,b
  for (int e = 0; e < 3; ++e) for (int j = 0; j < 12; ++j) ew[e][j] = in(q++);
  const float *lng = in(q++), *lnb = in(q++);
  const float* mw[8]; for (int j = 0; j < 8; ++j) mw[j] = in(q++);   // mixer l1w,l1b,l2w,l2b ×2
  const float* ocw[4]; for (int j = 0; j < 4; ++j) ocw[j] = in(q++); // out_cross

  // ---- workspace bump allocator ----
  char* wp = (char*)d_ws;
  auto alloc = [&](size_t n) {
    float* r = (float*)wp;
    wp += ((n * sizeof(float) + 255) / 256) * 256;
    return r;
  };
  float *seasonO[3], *trendLn[3];
  for (int i = 0; i < 3; ++i) { seasonO[i] = alloc(16L * Ts[i] * D_); trendLn[i] = alloc(16L * Ts[i] * D_); }
  float* mixed1 = alloc(16L * 128 * D_);
  float* mixed2 = alloc(16L * 64 * D_);
  const int Tm = 256, Fm = 129;
  float* season = alloc(16L * Tm * D_);
  float* trend  = alloc(16L * Tm * D_);
  float* re     = alloc(16L * Fm * D_);
  float* im     = alloc(16L * Fm * D_);
  float* amp    = alloc(16L * Fm * D_);
  float* thr    = alloc(16 * D_);
  float* cosF   = alloc((size_t)Fm * Tm);
  float* sinF   = alloc((size_t)Fm * Tm);
  float* icosT  = alloc((size_t)Tm * Fm);
  float* isinT  = alloc((size_t)Tm * Fm);
  int*   dims   = (int*)alloc(64);
  float* wts    = alloc(64);
  float* xpad   = alloc(16L * 2 * Tm * D_);
  float* mid    = alloc(16L * 2 * Tm * 256);
  float* cvo    = alloc(16L * 2 * Tm * D_);
  float* res    = alloc(16L * Tm * D_);
  float* moe    = alloc(16L * Tm * D_);
  float* feats  = alloc(16 * 1280);
  float* gh     = alloc(16 * 256);
  float* logits = alloc(64);
  float* wexp   = alloc(64);
  float* ctx    = alloc(16 * D_);
  float* wh     = alloc(16 * D_);
  float* wgate  = alloc(16);
  float* pm     = alloc(16L * 32 * D_);
  float* ih     = alloc(16L * 32 * 256);
  float* io     = alloc(16L * 32 * D_);
  float* ih2    = alloc(16L * 32 * 256);
  float* intraO = alloc(16L * Tm * D_);
  float* dT     = alloc(16L * D_ * 256);
  float* h1     = alloc(16L * D_ * D_);
  float* dO     = alloc(16L * D_ * D_);
  float* dback  = alloc(16L * D_ * D_);
  float* obuf   = alloc(16L * Tm * D_);
  float* ohid   = alloc(16L * Tm * 256);
  float* o2     = alloc(16L * Tm * D_);

  auto B1 = [](long n) { return dim3((unsigned)((n + 255) / 256)); };
  const int pss[3] = {8, 16, 32};

  for (int si = 0; si < 3; ++si) {
    const int T = Ts[si], F = T / 2 + 1;
    const long BTD = 16L * T * D_, BFD = 16L * F * D_;
    const float* x = X[si];

    // ---- DFT decomposition (rfft / mask / irfft as WMMA GEMMs) ----
    gen_dft_k<<<B1((long)F * T), 256, 0, stream>>>(cosF, sinF, icosT, isinT, T, F);
    gemm(stream, cosF, x, nullptr, re, F, D_, T, 16, 0, (long)T * D_, (long)F * D_, 0, 0);
    gemm(stream, sinF, x, nullptr, im, F, D_, T, 16, 0, (long)T * D_, (long)F * D_, 0, 0);
    amp_k<<<B1(BFD), 256, 0, stream>>>(re, im, amp, (int)BFD, F, D_);
    thr_k<<<B1(16 * D_), 256, 0, stream>>>(amp, thr, F, D_, 16);
    mask_k<<<B1(BFD), 256, 0, stream>>>(amp, thr, re, im, (int)BFD, F, D_);
    gemm(stream, icosT, re, nullptr, season, T, D_, F, 16, 0, (long)F * D_, (long)T * D_, 0, 0);
    gemm(stream, isinT, im, nullptr, season, T, D_, F, 16, 0, (long)F * D_, (long)T * D_, 0, 1);
    sub_k<<<B1(BTD), 256, 0, stream>>>(x, season, trend, (int)BTD);

    // ---- data-dependent periods + itcs softmax weights (on device) ----
    period_k<<<1, 256, 0, stream>>>(amp, dims, 16, F, D_, T);
    pw_k<<<1, 32, 0, stream>>>(amp, dims, wts, F, 16);

    // ---- ITCS (fused MCDC convs per period) ----
    fill0_k<<<B1(BTD), 256, 0, stream>>>(res, (int)BTD);
    for (int which = 0; which < 2; ++which) {
      pad_k<<<B1(16L * 2 * T * D_), 256, 0, stream>>>(season, xpad, dims, which, T, 16);
      mcdc_k<<<B1(16L * 2 * T * 256), 256, 0, stream>>>(xpad, c1w0, c1b0, c1w3, c1b3,
                                                        mid, dims, which, 16, D_, 256, 1);
      mcdc_k<<<B1(16L * 2 * T * D_), 256, 0, stream>>>(mid, c2w0, c2b0, c2w3, c2b3,
                                                       cvo, dims, which, 16, 256, D_, 0);
      accum_k<<<B1(BTD), 256, 0, stream>>>(cvo, wts, res, dims, which, T, 16);
    }
    sout_k<<<B1(BTD), 256, 0, stream>>>(res, season, seasonO[si], (int)BTD);

    // ---- IPMS (MoE over patch experts, all projections via WMMA) ----
    fill0_k<<<B1(BTD), 256, 0, stream>>>(moe, (int)BTD);
    feats_k<<<B1(16 * D_), 256, 0, stream>>>(trend, feats, T, 16);
    gemm(stream, feats, g1w, g1b, gh, 16, 256, 1280, 1, 0, 0, 0, 1, 0);
    gemm(stream, gh, g2w, g2b, logits, 16, 3, 256, 1, 0, 0, 0, 0, 0);
    moew_k<<<1, 32, 0, stream>>>(logits, wexp, 16);
    ctx_k<<<B1(16 * D_), 256, 0, stream>>>(feats, ctx, 16);
    for (int e = 0; e < 3; ++e) {
      int ps = pss[e], Np = T / ps;
      pmean_k<<<B1(16L * Np * D_), 256, 0, stream>>>(trend, pm, T, ps, 16);
      gemm(stream, pm,  ew[e][4], ew[e][5], ih, 16 * Np, 256, D_, 1, 0, 0, 0, 0, 0);
      gemm(stream, ih,  ew[e][6], ew[e][7], io, 16 * Np, D_, 256, 1, 0, 0, 0, 0, 0);
      gemm(stream, trend, ew[e][0], ew[e][1], ih2, 16 * Np, 256, ps * D_, 1, 0, 0, 0, 0, 0);
      gemm(stream, ih2, ew[e][2], ew[e][3], intraO, 16 * Np, ps * D_, 256, 1, 0, 0, 0, 0, 0);
      gemm(stream, ctx, ew[e][8], ew[e][9], wh, 16, D_, D_, 1, 0, 0, 0, 1, 0);
      gemm(stream, wh,  ew[e][10], ew[e][11], wgate, 16, 1, D_, 1, 0, 0, 0, 2, 0);
      fuse_k<<<B1(BTD), 256, 0, stream>>>(intraO, io, trend, wgate, wexp, moe, T, ps, e, 16);
    }
    ln_k<<<B1(16L * T), 256, 0, stream>>>(moe, trend, lng, lnb, trendLn[si], T, 16);
  }

  // ---- mixer over trend scales ----
  tr_k<<<B1(16L * 256 * D_), 256, 0, stream>>>(trendLn[0], dT, 256, D_, 16);
  gemm(stream, dT, mw[0], mw[1], h1, 16 * D_, 128, 256, 1, 0, 0, 0, 1, 0);
  gemm(stream, h1, mw[2], mw[3], dO, 16 * D_, 128, 128, 1, 0, 0, 0, 0, 0);
  tr_k<<<B1(16L * 128 * D_), 256, 0, stream>>>(dO, dback, D_, 128, 16);
  add_k<<<B1(16L * 128 * D_), 256, 0, stream>>>(trendLn[1], dback, mixed1, 16 * 128 * D_);
  tr_k<<<B1(16L * 128 * D_), 256, 0, stream>>>(mixed1, dT, 128, D_, 16);
  gemm(stream, dT, mw[4], mw[5], h1, 16 * D_, 64, 128, 1, 0, 0, 0, 1, 0);
  gemm(stream, h1, mw[6], mw[7], dO, 16 * D_, 64, 64, 1, 0, 0, 0, 0, 0);
  tr_k<<<B1(16L * D_ * 64), 256, 0, stream>>>(dO, dback, D_, 64, 16);
  add_k<<<B1(16L * 64 * D_), 256, 0, stream>>>(trendLn[2], dback, mixed2, 16 * 64 * D_);

  // ---- out_cross + residual, write outputs ----
  float* mix[3] = { trendLn[0], mixed1, mixed2 };
  long ooff = 0;
  for (int si = 0; si < 3; ++si) {
    int T = Ts[si]; long BTD = 16L * T * D_;
    add_k<<<B1(BTD), 256, 0, stream>>>(seasonO[si], mix[si], obuf, (int)BTD);
    gemm(stream, obuf, ocw[0], ocw[1], ohid, 16 * T, 256, D_, 1, 0, 0, 0, 1, 0);
    gemm(stream, ohid, ocw[2], ocw[3], o2, 16 * T, D_, 256, 1, 0, 0, 0, 0, 0);
    add_k<<<B1(BTD), 256, 0, stream>>>(X[si], o2, (float*)d_out + ooff, (int)BTD);
    ooff += BTD;
  }
}
